// CausalSelfAttention_90761248899693
// MI455X (gfx1250) — matrix operand order repaired and checked by reading, not compiled, not run on hardware
//
#include <hip/hip_runtime.h>
#include <hip/hip_bf16.h>
#include <stdint.h>

// MI455X / gfx1250 causal self-attention forward.
// - all matmuls via v_wmma_f32_16x16x32_bf16 (fp32 accumulate)
// - GEMM tiles staged into LDS by the Tensor Data Mover (tensor_load_to_lds),
//   double buffered on TENSORcnt so DMA overlaps WMMA
// - problem is L2-resident (~80MB traffic vs 192MB L2) -> matrix-engine bound

typedef __bf16 bf16;
typedef __bf16 bf16x16 __attribute__((ext_vector_type(16)));
typedef float  f32x8   __attribute__((ext_vector_type(8)));
typedef unsigned int u32x4 __attribute__((ext_vector_type(4)));
typedef int    i32x4  __attribute__((ext_vector_type(4)));
typedef int    i32x8  __attribute__((ext_vector_type(8)));

#define BATCH 2
#define TT    2048
#define CC    1024
#define NHEAD 16
#define HDIM  64
#define MTOT  (BATCH*TT)   // 4096

static __device__ __forceinline__ bf16 f2bf(float f) {
  unsigned u = __builtin_bit_cast(unsigned, f);
  unsigned r = u + 0x7FFFu + ((u >> 16) & 1u);   // round-to-nearest-even
  unsigned short h = (unsigned short)(r >> 16);
  return __builtin_bit_cast(bf16, h);
}

union FragU { bf16x16 v; u32x4 q[2]; };

static __device__ __forceinline__ bf16x16 ld_frag(const bf16* p0, const bf16* p1) {
  FragU u;
  u.q[0] = *(const u32x4*)p0;
  u.q[1] = *(const u32x4*)p1;
  return u.v;
}

// ------------------------------------------------------------------ TDM
// 2D tile load: 128 rows x 32 bf16 (64B) per row, row stride = stride elems,
// LDS rows padded +16B -> 80B pitch (matches [128][40] bf16 arrays).
static __device__ __forceinline__ void tdm_load_2d(unsigned lds_off,
                                                   const void* gptr,
                                                   unsigned row_stride_elems) {
  unsigned long long ga = (unsigned long long)gptr;
  u32x4 g0;
  g0[0] = 1u;                                  // count=1 (valid user descriptor)
  g0[1] = lds_off;                             // LDS byte address
  g0[2] = (unsigned)ga;                        // global_addr[95:64]
  g0[3] = (unsigned)(ga >> 32) | (2u << 30);   // addr hi | type=2 ("image")
  i32x8 g1;
  g1[0] = (int)((1u << 16) | (1u << 20) | (3u << 22) | (3u << 25));
          // data_size=2B | pad_enable | pad_interval=16dw(64B) | pad_amount=4dw(16B)
  g1[1] = (int)0xFFFF0000u;                    // tensor_dim0 lo16 (dim0=0x7FFFFFFF)
  g1[2] = (int)0xFFFF7FFFu;                    // dim0 hi16 | tensor_dim1 lo16
  g1[3] = (int)((32u << 16) | 0x7FFFu);        // dim1 hi16 | tile_dim0=32
  g1[4] = (int)128u;                           // tile_dim1=128 | tile_dim2=0
  g1[5] = (int)row_stride_elems;               // tensor_dim0_stride lo32
  g1[6] = 0;                                   // stride hi | dim1_stride lo
  g1[7] = 0;
  i32x4 zz = {0, 0, 0, 0};                     // groups 2/3: tile_dim3/4 = 0 (unused)
  asm volatile("tensor_load_to_lds %0, %1, %2, %3"
               :: "s"(g0), "s"(g1), "s"(zz), "s"(zz)
               : "memory");
}

// ---------------------------------------------------------------- converts
__global__ __launch_bounds__(256) void k_cvt_bf16(const float* __restrict__ in,
                                                  bf16* __restrict__ out, int n) {
  int i = blockIdx.x * 256 + threadIdx.x;
  if (i < n) out[i] = f2bf(in[i]);
}

// in: K x N row-major fp32  ->  out: N x K row-major bf16 (transpose)
__global__ __launch_bounds__(256) void k_cvt_bf16_T(const float* __restrict__ in,
                                                    bf16* __restrict__ out,
                                                    int K, int N) {
  int i = blockIdx.x * 256 + threadIdx.x;
  if (i < K * N) {
    int n = i / K, k = i - n * K;
    out[i] = f2bf(in[(size_t)k * N + n]);
  }
}

// ---------------------------------------------------------------- QKV GEMM
// xb: [4096][1024] bf16, WT: [3072][1024] bf16 (N-major), bias f32.
// Scatters into q/k [b,h,t,d] and vT [b,h,d,t] (bf16).
__global__ __launch_bounds__(256) void k_qkv_gemm(const bf16* __restrict__ xb,
                                                  const bf16* __restrict__ WT,
                                                  const float* __restrict__ bias,
                                                  bf16* __restrict__ qb,
                                                  bf16* __restrict__ kb,
                                                  bf16* __restrict__ vT) {
  __shared__ bf16 As[2][128][40];
  __shared__ bf16 Bs[2][128][40];
  const int tid = threadIdx.x;
  const int lane = tid & 31, wave = tid >> 5;
  const int waveM = wave >> 1, waveN = wave & 1;
  const int m0 = blockIdx.y * 128, n0 = blockIdx.x * 128;
  const int r = lane & 15, hw = lane >> 4;
  const unsigned ldsA = (unsigned)(uintptr_t)&As[0][0][0];
  const unsigned ldsB = (unsigned)(uintptr_t)&Bs[0][0][0];
  const bool issuer = (tid < 32);

  const f32x8 FZ = {0.f,0.f,0.f,0.f,0.f,0.f,0.f,0.f};
  f32x8 acc[2][4];
#pragma unroll
  for (int mi = 0; mi < 2; ++mi)
#pragma unroll
    for (int ni = 0; ni < 4; ++ni) acc[mi][ni] = FZ;

  if (issuer) {
    tdm_load_2d(ldsA, xb + (size_t)m0 * CC, CC);
    tdm_load_2d(ldsB, WT + (size_t)n0 * CC, CC);
  }

  int buf = 0;
  for (int k0 = 0; k0 < CC; k0 += 32) {
    __builtin_amdgcn_s_wait_tensorcnt(0);
    __syncthreads();                       // buffer `buf` ready for everyone
    if (issuer && (k0 + 32 < CC)) {        // prefetch next tile into other buffer
      tdm_load_2d(ldsA + (unsigned)(buf ^ 1) * 10240u, xb + (size_t)m0 * CC + k0 + 32, CC);
      tdm_load_2d(ldsB + (unsigned)(buf ^ 1) * 10240u, WT + (size_t)n0 * CC + k0 + 32, CC);
    }

    const bf16 (*Asb)[40] = As[buf];
    const bf16 (*Bsb)[40] = Bs[buf];
    bf16x16 a[2], b[4];
#pragma unroll
    for (int mi = 0; mi < 2; ++mi) {
      const bf16* base = &Asb[waveM * 32 + mi * 16 + r][hw * 8];
      a[mi] = ld_frag(base, base + 16);    // K chunks {kb..kb+8, kb+16..kb+24}
    }
#pragma unroll
    for (int ni = 0; ni < 4; ++ni) {
      const bf16* base = &Bsb[waveN * 64 + ni * 16 + r][hw * 8];
      b[ni] = ld_frag(base, base + 16);    // K chunks {kb..kb+8, kb+16..kb+24}
    }
#pragma unroll
    for (int mi = 0; mi < 2; ++mi)
#pragma unroll
      for (int ni = 0; ni < 4; ++ni)
        acc[mi][ni] = __builtin_amdgcn_wmma_f32_16x16x32_bf16(
            false, a[mi], false, b[ni], (short)0, acc[mi][ni], false, false);
    buf ^= 1;
  }

  // epilogue: `which` is block-uniform (128-wide tile never crosses 1024)
  const int which = n0 >> 10;              // 0=q, 1=k, 2=v
  if (which == 2) {
#pragma unroll
    for (int mi = 0; mi < 2; ++mi) {
#pragma unroll
      for (int ni = 0; ni < 4; ++ni) {
        int n = n0 + waveN * 64 + ni * 16 + r;
        float bv = bias[n];
        int rem = n & 1023, h = rem >> 6, d = rem & 63;
        int mbase = m0 + waveM * 32 + mi * 16 + hw * 8;
        int bb = mbase >> 11, t0 = mbase & 2047;
        union { u32x4 q; bf16 e[8]; } pk;
#pragma unroll
        for (int j = 0; j < 8; ++j) pk.e[j] = f2bf(acc[mi][ni][j] + bv);
        *(u32x4*)(vT + (((size_t)(bb * NHEAD + h) * HDIM + d) << 11) + t0) = pk.q;
      }
    }
  } else {
    bf16* __restrict__ dst = which ? kb : qb;
#pragma unroll
    for (int mi = 0; mi < 2; ++mi) {
#pragma unroll
      for (int ni = 0; ni < 4; ++ni) {
        int n = n0 + waveN * 64 + ni * 16 + r;
        float bv = bias[n];
        int rem = n & 1023, h = rem >> 6, d = rem & 63;
#pragma unroll
        for (int j = 0; j < 8; ++j) {
          int m = m0 + waveM * 32 + mi * 16 + hw * 8 + j;
          int bb = m >> 11, t = m & 2047;
          dst[(((size_t)(bb * NHEAD + h) * TT + t) << 6) + d] = f2bf(acc[mi][ni][j] + bv);
        }
      }
    }
  }
}

// ---------------------------------------------------------------- attention
// One 16-query tile per wave; keys in steps of 32; online softmax.
__global__ __launch_bounds__(256) void k_attn(const bf16* __restrict__ qb,
                                              const bf16* __restrict__ kb,
                                              const bf16* __restrict__ vT,
                                              bf16* __restrict__ y) {
  __shared__ bf16 Plds[8][16][40];
  const int tid = threadIdx.x, lane = tid & 31, wave = tid >> 5;
  const int gw = blockIdx.x * 8 + wave;     // 0..4095
  const int bh = gw >> 7;                   // 0..31  (batch*NH)
  const int q0 = (gw & 127) << 4;
  const int batch = bh >> 4, head = bh & 15;
  const int r = lane & 15, hw = lane >> 4;

  const bf16* qp = qb + (size_t)bh * TT * HDIM;
  const bf16* kp = kb + (size_t)bh * TT * HDIM;
  const bf16* vp = vT + (size_t)bh * HDIM * TT;

  bf16x16 qa[2];
#pragma unroll
  for (int kk = 0; kk < 2; ++kk) {
    const bf16* base = qp + (size_t)(q0 + r) * HDIM + kk * 32 + hw * 8;
    qa[kk] = ld_frag(base, base + 16);
  }

  const f32x8 FZ = {0.f,0.f,0.f,0.f,0.f,0.f,0.f,0.f};
  f32x8 o[4];
#pragma unroll
  for (int dt = 0; dt < 4; ++dt) o[dt] = FZ;
  float mr[8], lr[8];
#pragma unroll
  for (int j = 0; j < 8; ++j) { mr[j] = -1e30f; lr[j] = 0.f; }
  const float scale = 0.125f;   // 1/sqrt(64)

  for (int key0 = 0; key0 < q0 + 16; key0 += 32) {
    // S(16x32) = Q(16x64) @ K_tile^T(64x32)
    f32x8 s[2];
#pragma unroll
    for (int nt = 0; nt < 2; ++nt) {
      f32x8 z = FZ;
#pragma unroll
      for (int kk = 0; kk < 2; ++kk) {
        const bf16* base = kp + (size_t)(key0 + nt * 16 + r) * HDIM + kk * 32 + hw * 8;
        bf16x16 bk = ld_frag(base, base + 16);
        z = __builtin_amdgcn_wmma_f32_16x16x32_bf16(false, qa[kk], false, bk,
                                                    (short)0, z, false, false);
      }
      s[nt] = z;
    }

    // masked online softmax over 32 keys (row = hw*8 + j, cols on lanes)
    float p[2][8], rmax[8];
#pragma unroll
    for (int j = 0; j < 8; ++j) {
      int qi = q0 + hw * 8 + j;
      float v0 = s[0][j] * scale; if (key0 + r > qi)      v0 = -1e30f;
      float v1 = s[1][j] * scale; if (key0 + 16 + r > qi) v1 = -1e30f;
      p[0][j] = v0; p[1][j] = v1;
      float rm = fmaxf(v0, v1);
#pragma unroll
      for (int off = 1; off < 16; off <<= 1) rm = fmaxf(rm, __shfl_xor(rm, off, 32));
      rmax[j] = rm;
    }
#pragma unroll
    for (int j = 0; j < 8; ++j) {
      float mn = fmaxf(mr[j], rmax[j]);
      float alpha = __expf(mr[j] - mn);
      float e0 = __expf(p[0][j] - mn);
      float e1 = __expf(p[1][j] - mn);
      p[0][j] = e0; p[1][j] = e1;
      float rs = e0 + e1;
#pragma unroll
      for (int off = 1; off < 16; off <<= 1) rs += __shfl_xor(rs, off, 32);
      lr[j] = lr[j] * alpha + rs;
      mr[j] = mn;
#pragma unroll
      for (int dt = 0; dt < 4; ++dt) o[dt][j] *= alpha;
    }

    // C-layout -> A-layout for P via per-wave LDS slab
#pragma unroll
    for (int nt = 0; nt < 2; ++nt)
#pragma unroll
      for (int j = 0; j < 8; ++j)
        Plds[wave][hw * 8 + j][nt * 16 + r] = f2bf(p[nt][j]);
    const bf16* pb = &Plds[wave][r][hw * 8];
    bf16x16 pa = ld_frag(pb, pb + 16);

    // O(16x64) += P(16x32) @ V_tile(32x64); vT gives contiguous B-frag reads
#pragma unroll
    for (int dt = 0; dt < 4; ++dt) {
      const bf16* vb = vp + (size_t)(dt * 16 + r) * TT + key0 + hw * 8;
      bf16x16 vf = ld_frag(vb, vb + 16);
      o[dt] = __builtin_amdgcn_wmma_f32_16x16x32_bf16(false, pa, false, vf,
                                                      (short)0, o[dt], false, false);
    }
  }

  // normalize and write y[b,t,h*64+d] as bf16 (A-matrix of the proj GEMM)
#pragma unroll
  for (int dt = 0; dt < 4; ++dt) {
#pragma unroll
    for (int j = 0; j < 8; ++j) {
      float ov = o[dt][j] / lr[j];
      int t = q0 + hw * 8 + j;
      int c = head * HDIM + dt * 16 + r;
      y[((size_t)(batch * TT + t)) * CC + c] = f2bf(ov);
    }
  }
}

// ---------------------------------------------------------------- proj GEMM
// yb: [4096][1024] bf16, WT: [1024][1024] bf16 (N-major), out fp32.
__global__ __launch_bounds__(256) void k_proj_gemm(const bf16* __restrict__ yb,
                                                   const bf16* __restrict__ WT,
                                                   const float* __restrict__ bias,
                                                   float* __restrict__ out) {
  __shared__ bf16 As[2][128][40];
  __shared__ bf16 Bs[2][128][40];
  const int tid = threadIdx.x;
  const int lane = tid & 31, wave = tid >> 5;
  const int waveM = wave >> 1, waveN = wave & 1;
  const int m0 = blockIdx.y * 128, n0 = blockIdx.x * 128;
  const int r = lane & 15, hw = lane >> 4;
  const unsigned ldsA = (unsigned)(uintptr_t)&As[0][0][0];
  const unsigned ldsB = (unsigned)(uintptr_t)&Bs[0][0][0];
  const bool issuer = (tid < 32);

  const f32x8 FZ = {0.f,0.f,0.f,0.f,0.f,0.f,0.f,0.f};
  f32x8 acc[2][4];
#pragma unroll
  for (int mi = 0; mi < 2; ++mi)
#pragma unroll
    for (int ni = 0; ni < 4; ++ni) acc[mi][ni] = FZ;

  if (issuer) {
    tdm_load_2d(ldsA, yb + (size_t)m0 * CC, CC);
    tdm_load_2d(ldsB, WT + (size_t)n0 * CC, CC);
  }

  int buf = 0;
  for (int k0 = 0; k0 < CC; k0 += 32) {
    __builtin_amdgcn_s_wait_tensorcnt(0);
    __syncthreads();
    if (issuer && (k0 + 32 < CC)) {
      tdm_load_2d(ldsA + (unsigned)(buf ^ 1) * 10240u, yb + (size_t)m0 * CC + k0 + 32, CC);
      tdm_load_2d(ldsB + (unsigned)(buf ^ 1) * 10240u, WT + (size_t)n0 * CC + k0 + 32, CC);
    }

    const bf16 (*Asb)[40] = As[buf];
    const bf16 (*Bsb)[40] = Bs[buf];
    bf16x16 a[2], b[4];
#pragma unroll
    for (int mi = 0; mi < 2; ++mi) {
      const bf16* base = &Asb[waveM * 32 + mi * 16 + r][hw * 8];
      a[mi] = ld_frag(base, base + 16);
    }
#pragma unroll
    for (int ni = 0; ni < 4; ++ni) {
      const bf16* base = &Bsb[waveN * 64 + ni * 16 + r][hw * 8];
      b[ni] = ld_frag(base, base + 16);
    }
#pragma unroll
    for (int mi = 0; mi < 2; ++mi)
#pragma unroll
      for (int ni = 0; ni < 4; ++ni)
        acc[mi][ni] = __builtin_amdgcn_wmma_f32_16x16x32_bf16(
            false, a[mi], false, b[ni], (short)0, acc[mi][ni], false, false);
    buf ^= 1;
  }

#pragma unroll
  for (int mi = 0; mi < 2; ++mi) {
#pragma unroll
    for (int ni = 0; ni < 4; ++ni) {
      int n = n0 + waveN * 64 + ni * 16 + r;
      float bv = bias[n];
#pragma unroll
      for (int j = 0; j < 8; ++j) {
        int m = m0 + waveM * 32 + mi * 16 + hw * 8 + j;
        out[(size_t)m * CC + n] = acc[mi][ni][j] + bv;
      }
    }
  }
}

// ---------------------------------------------------------------- launcher
extern "C" void kernel_launch(void* const* d_in, const int* in_sizes, int n_in,
                              void* d_out, int out_size, void* d_ws, size_t ws_size,
                              hipStream_t stream) {
  const float* x     = (const float*)d_in[0];   // (2,2048,1024)
  const float* Wqkv  = (const float*)d_in[1];   // (1024,3072)
  const float* bqkv  = (const float*)d_in[2];   // (3072,)
  const float* Wproj = (const float*)d_in[3];   // (1024,1024)
  const float* bproj = (const float*)d_in[4];   // (1024,)
  float* out = (float*)d_out;

  char* ws = (char*)d_ws;
  // bf16 workspace layout (bytes), total 48 MiB
  bf16* xb     = (bf16*)(ws);                        //  8 MiB : x bf16
  bf16* WqkvT  = (bf16*)(ws + (((size_t)8)  << 20)); //  6 MiB : Wqkv^T bf16
  bf16* WprojT = (bf16*)(ws + (((size_t)14) << 20)); //  2 MiB : Wproj^T bf16
  bf16* qb     = (bf16*)(ws + (((size_t)16) << 20)); //  8 MiB : q [b,h,t,d]
  bf16* kb     = (bf16*)(ws + (((size_t)24) << 20)); //  8 MiB : k [b,h,t,d]
  bf16* vT     = (bf16*)(ws + (((size_t)32) << 20)); //  8 MiB : v [b,h,d,t]
  bf16* yb     = (bf16*)(ws + (((size_t)40) << 20)); //  8 MiB : attn out [b,t,c]

  const int nx = MTOT * CC;          // 4194304
  k_cvt_bf16<<<dim3((nx + 255) / 256), dim3(256), 0, stream>>>(x, xb, nx);
  k_cvt_bf16_T<<<dim3((CC * 3 * CC + 255) / 256), dim3(256), 0, stream>>>(Wqkv, WqkvT, CC, 3 * CC);
  k_cvt_bf16_T<<<dim3((CC * CC + 255) / 256), dim3(256), 0, stream>>>(Wproj, WprojT, CC, CC);

  k_qkv_gemm<<<dim3(24, 32), dim3(256), 0, stream>>>(xb, WqkvT, bqkv, qb, kb, vT);
  k_attn<<<dim3(512), dim3(256), 0, stream>>>(qb, kb, vT, yb);
  k_proj_gemm<<<dim3(8, 32), dim3(256), 0, stream>>>(yb, WprojT, bproj, out);
}
